// LSTM_24850680774748
// MI455X (gfx1250) — compile-verified, matrix operand order checked by reading, not run on hardware
//
#include <hip/hip_runtime.h>
#include <hip/hip_bf16.h>
#include <math.h>

typedef __attribute__((ext_vector_type(16))) _Float16 v16h;
typedef __attribute__((ext_vector_type(8)))  float    v8f;

#define HIDDEN   80
#define G4       (4 * HIDDEN)      // 320 gate rows
#define KPAD     96                // 80 padded to 3 k-tiles of 32
#define NWAVE    20                // 320 / 16 rows per wave
#define NTHREADS (NWAVE * 32)      // 640
#define TSTEPS   8192

union BU { v16h v; unsigned int u[8]; };

// Map A/B register element e (0..15 of a v16h) + lane-half g to its K index
// within a 32-wide k-tile, per CDNA5 16-bit A-matrix 16x32 layout.
__device__ __forceinline__ int k_of(int e, int g) {
    int v  = e >> 1;
    int hi = e & 1;
    return ((v < 4) ? 0 : 16) + (g ? 8 : 0) + 2 * (v & 3) + hi;
}

// Fast saturating sigmoid / tanh: __expf (v_exp path) + v_rcp_f32.
// exp overflow -> inf -> rcp -> 0, so both saturate correctly at the tails.
__device__ __forceinline__ float fast_sigmoid(float x) {
    return __builtin_amdgcn_rcpf(1.0f + __expf(-x));
}
__device__ __forceinline__ float fast_tanh(float x) {
    return 1.0f - 2.0f * __builtin_amdgcn_rcpf(1.0f + __expf(2.0f * x));
}

__global__ void __launch_bounds__(NTHREADS)
lstm_scan_wmma(const float* __restrict__ x,
               const float* __restrict__ w_ih,   // (320,1)
               const float* __restrict__ w_hh,   // (320,80) row-major
               const float* __restrict__ b_ih,   // (320)
               const float* __restrict__ b_hh,   // (320)
               const float* __restrict__ fc1_w,  // (64,80)
               const float* __restrict__ fc1_b,  // (64)
               const float* __restrict__ fc2_w,  // (32,64)
               const float* __restrict__ fc2_b,  // (32)
               const float* __restrict__ fc3_w,  // (1,32)
               const float* __restrict__ fc3_b,  // (1)
               float* __restrict__ out,
               int T)
{
    __shared__ float        xs[TSTEPS];        // 32 KB: whole input sequence
    __shared__ float        gates[G4];         // per-step gate pre-activations
    __shared__ float        bsum[G4];          // b_ih + b_hh
    __shared__ float        wih[G4];           // w_ih column (input size 1)
    __shared__ unsigned int h16u[KPAD / 2];    // h as packed fp16 pairs, zero-padded K=80..95
    __shared__ float        h32[HIDDEN];       // fp32 h for the MLP head
    __shared__ float        y1[64];
    __shared__ float        y2[32];

    const int tid  = threadIdx.x;
    const int lane = tid & 31;
    const int wave = tid >> 5;           // 0..19 -> gate-row tile
    const int g    = (lane >= 16);       // lane half selects K sub-group
    const int mrow = wave * 16 + (lane & 15);

    // ---- one-time staging ----
    for (int i = tid; i < T; i += NTHREADS)        xs[i]   = x[i];
    for (int i = tid; i < G4; i += NTHREADS) {
        bsum[i] = b_ih[i] + b_hh[i];
        wih[i]  = w_ih[i];
    }
    for (int i = tid; i < KPAD / 2; i += NTHREADS) h16u[i] = 0u;  // h0 = 0 (+ K padding)

    // ---- load this wave's 16 x 96 fp16 weight slice into A registers ----
    v16h A[3];
    #pragma unroll
    for (int kt = 0; kt < 3; ++kt) {
        #pragma unroll
        for (int e = 0; e < 16; ++e) {
            int K = 32 * kt + k_of(e, g);
            A[kt][e] = (K < HIDDEN) ? (_Float16)w_hh[mrow * HIDDEN + K] : (_Float16)0.0f;
        }
    }

    float c = 0.0f;                      // cell state, register-resident (threads 0..79)
    __syncthreads();

    // ---- sequential LSTM scan ----
    for (int t = 0; t < T; ++t) {
        // Broadcast h (fp16 pairs from LDS) into the B operand: every column = h,
        // so all columns of D equal W_tile @ h.
        BU b[3];
        #pragma unroll
        for (int kt = 0; kt < 3; ++kt) {
            #pragma unroll
            for (int vv = 0; vv < 8; ++vv) {
                int idx = 16 * kt + ((vv < 4) ? 0 : 8) + (g ? 4 : 0) + (vv & 3);
                b[kt].u[vv] = h16u[idx];
            }
        }

        v8f acc = {};
        acc = __builtin_amdgcn_wmma_f32_16x16x32_f16(false, A[0], false, b[0].v,
                                                     (short)0, acc, false, false);
        acc = __builtin_amdgcn_wmma_f32_16x16x32_f16(false, A[1], false, b[1].v,
                                                     (short)0, acc, false, false);
        acc = __builtin_amdgcn_wmma_f32_16x16x32_f16(false, A[2], false, b[2].v,
                                                     (short)0, acc, false, false);

        // D layout: vgpr v, lane l -> row (v + 8*(l>=16)), columns identical.
        if ((lane & 15) == 0) {
            int   mbase = wave * 16 + (g ? 8 : 0);
            float xt    = xs[t];
            #pragma unroll
            for (int v = 0; v < 8; ++v) {
                int r = mbase + v;
                gates[r] = acc[v] + bsum[r] + wih[r] * xt;
            }
        }
        __syncthreads();

        // Elementwise LSTM cell update (fp32), threads 0..79.
        if (tid < HIDDEN) {
            float gi = gates[tid];
            float gf = gates[HIDDEN + tid];
            float gg = gates[2 * HIDDEN + tid];
            float go = gates[3 * HIDDEN + tid];
            float si = fast_sigmoid(gi);
            float sf = fast_sigmoid(gf);
            float tg = fast_tanh(gg);
            float so = fast_sigmoid(go);
            c = sf * c + si * tg;
            float h = so * fast_tanh(c);
            h32[tid] = h;
            ((_Float16*)h16u)[tid] = (_Float16)h;   // fp16 copy feeding next step's B
        }
        __syncthreads();
    }

    // ---- MLP head: 80 -> 64 (relu) -> 32 (relu) -> 1 ----
    if (tid < 64) {
        float a = fc1_b[tid];
        for (int k = 0; k < HIDDEN; ++k) a += h32[k] * fc1_w[tid * HIDDEN + k];
        y1[tid] = a > 0.0f ? a : 0.0f;
    }
    __syncthreads();
    if (tid < 32) {
        float a = fc2_b[tid];
        for (int k = 0; k < 64; ++k) a += y1[k] * fc2_w[tid * 64 + k];
        y2[tid] = a > 0.0f ? a : 0.0f;
    }
    __syncthreads();
    if (tid == 0) {
        float a = fc3_b[0];
        for (int k = 0; k < 32; ++k) a += y2[k] * fc3_w[k];
        out[0] = a;
    }
}

extern "C" void kernel_launch(void* const* d_in, const int* in_sizes, int n_in,
                              void* d_out, int out_size, void* d_ws, size_t ws_size,
                              hipStream_t stream) {
    const float* x     = (const float*)d_in[0];
    const float* w_ih  = (const float*)d_in[1];
    const float* w_hh  = (const float*)d_in[2];
    const float* b_ih  = (const float*)d_in[3];
    const float* b_hh  = (const float*)d_in[4];
    const float* fc1_w = (const float*)d_in[5];
    const float* fc1_b = (const float*)d_in[6];
    const float* fc2_w = (const float*)d_in[7];
    const float* fc2_b = (const float*)d_in[8];
    const float* fc3_w = (const float*)d_in[9];
    const float* fc3_b = (const float*)d_in[10];
    float* out = (float*)d_out;

    int T = in_sizes[0];   // INPUT_SIZE == 1, so flat count == T (8192)
    if (T > TSTEPS) T = TSTEPS;

    lstm_scan_wmma<<<1, NTHREADS, 0, stream>>>(x, w_ih, w_hh, b_ih, b_hh,
                                               fc1_w, fc1_b, fc2_w, fc2_b,
                                               fc3_w, fc3_b, out, T);
}
